// MoELayer_54348516163746
// MI455X (gfx1250) — matrix operand order repaired and checked by reading, not compiled, not run on hardware
//
#include <hip/hip_runtime.h>
#include <math.h>

// Problem constants (match reference)
#define NTOK 16384
#define DDIM 1024
#define NE   8
#define HDIM 2048
#define ODIM 1024
#define NSLOT (2 * NTOK)   // TOP_K = 2 -> exactly 2N routed slots

typedef __bf16 bf16_t;
typedef __attribute__((ext_vector_type(16))) __bf16 v16bf;
typedef __attribute__((ext_vector_type(8)))  __bf16 v8bf;
typedef __attribute__((ext_vector_type(4)))  __bf16 v4bf;
typedef __attribute__((ext_vector_type(2)))  __bf16 v2bf;
typedef __attribute__((ext_vector_type(8)))  float  v8f;

// ---------------------------------------------------------------------------
// LDS tile layouts (every WMMA fragment = two contiguous 16B ds_load_b128):
//   A: row-major [row][LDA], lane reads K kh..kh+7 and 16+kh..16+kh+7
//   B: COLUMN-major [col][LDBT], lane reads K kh..kh+7 and 16+kh..16+kh+7
// ---------------------------------------------------------------------------
#define LDA  40   // 32 + 8 pad (80B row stride, 16B aligned)
#define LDBT 40   // 32 + 8 pad (80B col stride, 16B aligned)

__device__ __forceinline__ v16bf frag_a(const bf16_t* sA, int row0) {
  const int lane = threadIdx.x & 31;
  const bf16_t* p = sA + (row0 + (lane & 15)) * LDA + ((lane >> 4) * 8);
  const v8bf lo = *(const v8bf*)p;          // K = kh .. kh+7
  const v8bf hi = *(const v8bf*)(p + 16);   // K = 16+kh .. 16+kh+7
  return __builtin_shufflevector(lo, hi, 0, 1, 2, 3, 4, 5, 6, 7,
                                 8, 9, 10, 11, 12, 13, 14, 15);
}

__device__ __forceinline__ v16bf frag_b(const bf16_t* sB, int col0) {
  const int lane = threadIdx.x & 31;
  const bf16_t* p = sB + (col0 + (lane & 15)) * LDBT + ((lane >> 4) * 8);
  const v8bf lo = *(const v8bf*)p;          // K = kh .. kh+7
  const v8bf hi = *(const v8bf*)(p + 16);   // K = 16+kh .. 16+kh+7
  return __builtin_shufflevector(lo, hi, 0, 1, 2, 3, 4, 5, 6, 7,
                                 8, 9, 10, 11, 12, 13, 14, 15);
}

// ---------------------------------------------------------------------------
// Gate: logits = tanh(x @ Wg1) @ Wg2 ; softmax ; top-2 ; renormalize.
// One wave per token: coalesced x reads, 16 partial dots, shfl reduction.
// ---------------------------------------------------------------------------
__global__ __launch_bounds__(256)
void gate_kernel(const float* __restrict__ x, const float* __restrict__ Wg1,
                 const float* __restrict__ Wg2,
                 int* __restrict__ tok_e, float* __restrict__ tok_w,
                 int* __restrict__ counts)
{
  const int lane = threadIdx.x & 31;
  const int n = blockIdx.x * 8 + (threadIdx.x >> 5);
  float acc[16];
#pragma unroll
  for (int j = 0; j < 16; ++j) acc[j] = 0.f;
  for (int d = lane; d < DDIM; d += 32) {
    const float xv = x[(size_t)n * DDIM + d];
    const float4* wr = (const float4*)(Wg1 + d * 16);
#pragma unroll
    for (int q = 0; q < 4; ++q) {
      const float4 w = wr[q];
      acc[q * 4 + 0] = fmaf(xv, w.x, acc[q * 4 + 0]);
      acc[q * 4 + 1] = fmaf(xv, w.y, acc[q * 4 + 1]);
      acc[q * 4 + 2] = fmaf(xv, w.z, acc[q * 4 + 2]);
      acc[q * 4 + 3] = fmaf(xv, w.w, acc[q * 4 + 3]);
    }
  }
#pragma unroll
  for (int j = 0; j < 16; ++j)
    for (int m = 16; m >= 1; m >>= 1)
      acc[j] += __shfl_xor(acc[j], m, 32);

  if (lane == 0) {
    float logit[8];
#pragma unroll
    for (int e = 0; e < 8; ++e) logit[e] = 0.f;
#pragma unroll
    for (int j = 0; j < 16; ++j) {
      const float tv = tanhf(acc[j]);
#pragma unroll
      for (int e = 0; e < 8; ++e)
        logit[e] = fmaf(tv, Wg2[j * 8 + e], logit[e]);
    }
    float mx = logit[0];
#pragma unroll
    for (int e = 1; e < 8; ++e) mx = fmaxf(mx, logit[e]);
    float p[8], sum = 0.f;
#pragma unroll
    for (int e = 0; e < 8; ++e) { p[e] = expf(logit[e] - mx); sum += p[e]; }
    const float rs = 1.f / sum;
#pragma unroll
    for (int e = 0; e < 8; ++e) p[e] *= rs;
    int i0 = 0;
#pragma unroll
    for (int e = 1; e < 8; ++e) if (p[e] > p[i0]) i0 = e;
    int i1 = (i0 == 0) ? 1 : 0;
#pragma unroll
    for (int e = 0; e < 8; ++e) if (e != i0 && p[e] > p[i1]) i1 = e;
    const float denom = p[i0] + p[i1] + 1e-12f;
    tok_e[2 * n]     = i0; tok_w[2 * n]     = p[i0] / denom;
    tok_e[2 * n + 1] = i1; tok_w[2 * n + 1] = p[i1] / denom;
    atomicAdd(&counts[i0], 1);
    atomicAdd(&counts[i1], 1);
  }
}

__global__ void zero_counts_kernel(int* __restrict__ counts) {
  if (threadIdx.x < NE) counts[threadIdx.x] = 0;
}

__global__ void prefix_kernel(const int* __restrict__ counts,
                              int* __restrict__ offsets,
                              int* __restrict__ cursors) {
  if (threadIdx.x == 0) {
    int run = 0;
    for (int e = 0; e < NE; ++e) { offsets[e] = run; cursors[e] = run; run += counts[e]; }
  }
}

__global__ __launch_bounds__(256)
void scatter_kernel(const int* __restrict__ tok_e, int* __restrict__ cursors,
                    int* __restrict__ slot_token, int* __restrict__ slot_expert,
                    int* __restrict__ tok_slot)
{
  const int n = blockIdx.x * 256 + threadIdx.x;
#pragma unroll
  for (int k = 0; k < 2; ++k) {
    const int e = tok_e[2 * n + k];
    const int s = atomicAdd(&cursors[e], 1);
    slot_token[s]  = n;
    slot_expert[s] = e;
    tok_slot[2 * n + k] = s;
  }
}

// ---------------------------------------------------------------------------
// Expert GEMM: 128x128x32 tiles, 8 waves (4x2), each wave 2x4 WMMA fragments.
// Software-pipelined: tile k+1 global loads issued before tile k's WMMA pass.
// Row gather pointers hoisted out of the K loop; OOB rows are clamped (their
// results are discarded by the guarded epilogue), so no predication needed.
// MODE 0: A = gathered x rows (fp32 -> bf16), out = h bf16 (+bias)
// MODE 1: A = bf16 activation rows,          out = h bf16 (+bias)
// MODE 2: A = bf16 activation rows,          out = y fp32 (+bias)
// ---------------------------------------------------------------------------
template <int MODE>
__global__ __launch_bounds__(256)
void moe_gemm_kernel(const float* __restrict__ Xf, const bf16_t* __restrict__ Ab,
                     const float* __restrict__ Wbase, const float* __restrict__ bias,
                     bf16_t* __restrict__ Hout, float* __restrict__ Yout,
                     const int* __restrict__ counts, const int* __restrict__ offsets,
                     const int* __restrict__ slot_token, int Ktot, int Nout)
{
  const int e   = blockIdx.z;
  const int cnt = counts[e];
  const int m0  = blockIdx.y * 128;
  if (m0 >= cnt) return;
  const int seg = offsets[e];
  const int n0  = blockIdx.x * 128;

  __shared__ bf16_t sA[128 * LDA];   // row-major
  __shared__ bf16_t sB[128 * LDBT];  // column-major

  const int t   = threadIdx.x;
  const int wid = t >> 5;
  const int wm  = (wid & 3) * 32;
  const int wn  = (wid >> 2) * 64;

  const v8f zero8 = {0.f, 0.f, 0.f, 0.f, 0.f, 0.f, 0.f, 0.f};
  v8f acc[2][4];
#pragma unroll
  for (int i = 0; i < 2; ++i)
#pragma unroll
    for (int j = 0; j < 4; ++j) acc[i][j] = zero8;

  const float* Wexp = Wbase + (size_t)e * Ktot * Nout;

  // ---- hoisted per-thread source pointers (fixed across the K loop) ----
  // A staging: each thread covers 4 rows (r, r+32, r+64, r+96) x 4 K elems
  const int ar  = t >> 3;
  const int akk = (t & 7) * 4;
  const float*  aSrcF[4];
  const bf16_t* aSrcB[4];
#pragma unroll
  for (int p = 0; p < 4; ++p) {
    const int ml = min(m0 + ar + p * 32, cnt - 1);   // clamp: results discarded
    if (MODE == 0) {
      const int tok = slot_token[seg + ml];
      aSrcF[p] = Xf + (size_t)tok * DDIM + akk;
    } else {
      aSrcB[p] = Ab + (size_t)(seg + ml) * Ktot + akk;
    }
  }
  // B staging: each thread covers a 2(K) x 8(N) patch
  const int bkp = (t >> 4) * 2;
  const int bc0 = (t & 15) * 8;
  const float* bSrc = Wexp + (size_t)bkp * Nout + n0 + bc0;

  // ---- pipeline registers ----
  float4 aRegF[4];
  v4bf   aRegB[4];
  float4 bReg[4];
  auto load_globals = [&](int kb) {
#pragma unroll
    for (int p = 0; p < 4; ++p) {
      if (MODE == 0) aRegF[p] = *(const float4*)(aSrcF[p] + kb);
      else           aRegB[p] = *(const v4bf*)(aSrcB[p] + kb);
    }
    const float* s0 = bSrc + (size_t)kb * Nout;
    bReg[0] = *(const float4*)(s0);
    bReg[1] = *(const float4*)(s0 + 4);
    bReg[2] = *(const float4*)(s0 + Nout);
    bReg[3] = *(const float4*)(s0 + Nout + 4);
  };

  load_globals(0);
  for (int kb = 0; kb < Ktot; kb += 32) {
    // ---- stage registers -> LDS ----
#pragma unroll
    for (int p = 0; p < 4; ++p) {
      if (MODE == 0) {
        const v4bf q = {(bf16_t)aRegF[p].x, (bf16_t)aRegF[p].y,
                        (bf16_t)aRegF[p].z, (bf16_t)aRegF[p].w};
        *(v4bf*)&sA[(ar + p * 32) * LDA + akk] = q;
      } else {
        *(v4bf*)&sA[(ar + p * 32) * LDA + akk] = aRegB[p];
      }
    }
    {
      const float va[8] = {bReg[0].x, bReg[0].y, bReg[0].z, bReg[0].w,
                           bReg[1].x, bReg[1].y, bReg[1].z, bReg[1].w};
      const float vb[8] = {bReg[2].x, bReg[2].y, bReg[2].z, bReg[2].w,
                           bReg[3].x, bReg[3].y, bReg[3].z, bReg[3].w};
#pragma unroll
      for (int j = 0; j < 8; ++j) {
        const v2bf pk = {(bf16_t)va[j], (bf16_t)vb[j]};
        *(v2bf*)&sB[(bc0 + j) * LDBT + bkp] = pk;   // packed (k, k+1) pair
      }
    }
    __syncthreads();

    // issue next tile's global loads before this tile's WMMA pass
    if (kb + 32 < Ktot) load_globals(kb + 32);
    if (kb + 64 < Ktot)
      __builtin_prefetch(bSrc + (size_t)(kb + 64) * Nout, 0, 1);

    v16bf afr[2], bfr[4];
#pragma unroll
    for (int i = 0; i < 2; ++i) afr[i] = frag_a(sA, wm + i * 16);
#pragma unroll
    for (int j = 0; j < 4; ++j) bfr[j] = frag_b(sB, wn + j * 16);
#pragma unroll
    for (int i = 0; i < 2; ++i)
#pragma unroll
      for (int j = 0; j < 4; ++j)
        acc[i][j] = __builtin_amdgcn_wmma_f32_16x16x32_bf16(
            false, afr[i], false, bfr[j], (short)0, acc[i][j], false, false);
    __syncthreads();
  }

  // epilogue (C/D layout: VGPR r -> M = r + 8*(lane>=16), N = lane&15)
  const int lane = t & 31;
  const int nl = lane & 15;
  const int mh = (lane >> 4) * 8;
#pragma unroll
  for (int i = 0; i < 2; ++i) {
#pragma unroll
    for (int j = 0; j < 4; ++j) {
      const int col = n0 + wn + j * 16 + nl;
      const float bv = bias[(size_t)e * Nout + col];
#pragma unroll
      for (int r = 0; r < 8; ++r) {
        const int rl = m0 + wm + i * 16 + mh + r;
        if (rl < cnt) {
          const float val = acc[i][j][r] + bv;
          if (MODE == 2)
            Yout[(size_t)(seg + rl) * Nout + col] = val;
          else
            Hout[(size_t)(seg + rl) * Nout + col] = (bf16_t)val;
        }
      }
    }
  }
}

// ---------------------------------------------------------------------------
// Fused LayerNorm (per expert gamma/beta) + exact GELU, in-place on bf16 rows.
// One 256-thread block per slot row (H = 2048 -> 8 elems/thread).
// ---------------------------------------------------------------------------
__global__ __launch_bounds__(256)
void ln_gelu_kernel(bf16_t* __restrict__ h, const float* __restrict__ gam,
                    const float* __restrict__ bet, const int* __restrict__ slot_expert)
{
  const int slot = blockIdx.x;
  const int e = slot_expert[slot];
  bf16_t* row = h + (size_t)slot * HDIM;
  const int t = threadIdx.x;
  float v[8], s = 0.f, s2 = 0.f;
#pragma unroll
  for (int i = 0; i < 8; ++i) {
    const float xv = (float)row[t + i * 256];
    v[i] = xv; s += xv; s2 += xv * xv;
  }
  __shared__ float sh[256], sh2[256];
  sh[t] = s; sh2[t] = s2;
  __syncthreads();
  for (int off = 128; off > 0; off >>= 1) {
    if (t < off) { sh[t] += sh[t + off]; sh2[t] += sh2[t + off]; }
    __syncthreads();
  }
  const float mean = sh[0] * (1.f / HDIM);
  const float var  = sh2[0] * (1.f / HDIM) - mean * mean;
  const float inv  = rsqrtf(var + 1e-5f);
#pragma unroll
  for (int i = 0; i < 8; ++i) {
    const int c = t + i * 256;
    const float xn = (v[i] - mean) * inv * gam[(size_t)e * HDIM + c] + bet[(size_t)e * HDIM + c];
    const float ge = 0.5f * xn * (1.f + erff(xn * 0.70710678118654752f));
    row[c] = (bf16_t)ge;
  }
}

// ---------------------------------------------------------------------------
// Deterministic top-2 weighted combine: out[n] = w0*y[s0] + w1*y[s1]
// ---------------------------------------------------------------------------
__global__ __launch_bounds__(256)
void combine_kernel(const float* __restrict__ y, const int* __restrict__ tok_slot,
                    const float* __restrict__ tok_w, float* __restrict__ out)
{
  const int O4 = ODIM / 4;
  const int idx = blockIdx.x * 256 + threadIdx.x;   // over N * O4
  const int n = idx / O4;
  const int c = idx - n * O4;
  const int s0 = tok_slot[2 * n], s1 = tok_slot[2 * n + 1];
  const float w0 = tok_w[2 * n], w1 = tok_w[2 * n + 1];
  const float4 a = ((const float4*)y)[(size_t)s0 * O4 + c];
  const float4 b = ((const float4*)y)[(size_t)s1 * O4 + c];
  float4 o;
  o.x = w0 * a.x + w1 * b.x;
  o.y = w0 * a.y + w1 * b.y;
  o.z = w0 * a.z + w1 * b.z;
  o.w = w0 * a.w + w1 * b.w;
  ((float4*)out)[(size_t)n * O4 + c] = o;
}

// ---------------------------------------------------------------------------
extern "C" void kernel_launch(void* const* d_in, const int* in_sizes, int n_in,
                              void* d_out, int out_size, void* d_ws, size_t ws_size,
                              hipStream_t stream)
{
  const float* x   = (const float*)d_in[0];
  const float* Wg1 = (const float*)d_in[1];
  const float* Wg2 = (const float*)d_in[2];
  const float* W1  = (const float*)d_in[3];
  const float* b1  = (const float*)d_in[4];
  const float* g1  = (const float*)d_in[5];
  const float* be1 = (const float*)d_in[6];
  const float* W2  = (const float*)d_in[7];
  const float* b2  = (const float*)d_in[8];
  const float* g2  = (const float*)d_in[9];
  const float* be2 = (const float*)d_in[10];
  const float* W3  = (const float*)d_in[11];
  const float* b3  = (const float*)d_in[12];
  float* out = (float*)d_out;

  // Workspace carve-up (~275 MB): routing (~1MB) + h1/h2 bf16 (128MB each);
  // y (fp32, 128MB) aliases h1 which is dead by GEMM3.
  char* ws = (char*)d_ws;
  size_t off = 0;
  auto carve = [&](size_t bytes, size_t align) -> char* {
    off = (off + align - 1) & ~(align - 1);
    char* p = ws + off; off += bytes; return p;
  };
  int*   counts      = (int*)  carve(NE * sizeof(int), 256);
  int*   offsets     = (int*)  carve(NE * sizeof(int), 256);
  int*   cursors     = (int*)  carve(NE * sizeof(int), 256);
  int*   tok_e       = (int*)  carve((size_t)NSLOT * sizeof(int), 256);
  float* tok_w       = (float*)carve((size_t)NSLOT * sizeof(float), 256);
  int*   tok_slot    = (int*)  carve((size_t)NSLOT * sizeof(int), 256);
  int*   slot_token  = (int*)  carve((size_t)NSLOT * sizeof(int), 256);
  int*   slot_expert = (int*)  carve((size_t)NSLOT * sizeof(int), 256);
  bf16_t* h1 = (bf16_t*)carve((size_t)NSLOT * HDIM * sizeof(bf16_t), 256);
  bf16_t* h2 = (bf16_t*)carve((size_t)NSLOT * HDIM * sizeof(bf16_t), 256);
  float*  yb = (float*)h1;   // alias: NSLOT*ODIM*4 == NSLOT*HDIM*2 bytes

  zero_counts_kernel<<<1, 32, 0, stream>>>(counts);
  gate_kernel<<<NTOK / 8, 256, 0, stream>>>(x, Wg1, Wg2, tok_e, tok_w, counts);
  prefix_kernel<<<1, 32, 0, stream>>>(counts, offsets, cursors);
  scatter_kernel<<<NTOK / 256, 256, 0, stream>>>(tok_e, cursors, slot_token,
                                                 slot_expert, tok_slot);

  dim3 gH(HDIM / 128, NTOK / 128, NE);   // N tiles x worst-case M tiles x experts
  dim3 gO(ODIM / 128, NTOK / 128, NE);

  moe_gemm_kernel<0><<<gH, 256, 0, stream>>>(x, nullptr, W1, b1, h1, nullptr,
                                             counts, offsets, slot_token, DDIM, HDIM);
  ln_gelu_kernel<<<NSLOT, 256, 0, stream>>>(h1, g1, be1, slot_expert);
  moe_gemm_kernel<1><<<gH, 256, 0, stream>>>(nullptr, h1, W2, b2, h2, nullptr,
                                             counts, offsets, slot_token, HDIM, HDIM);
  ln_gelu_kernel<<<NSLOT, 256, 0, stream>>>(h2, g2, be2, slot_expert);
  moe_gemm_kernel<2><<<gO, 256, 0, stream>>>(nullptr, h2, W3, b3, nullptr, yb,
                                             counts, offsets, slot_token, HDIM, ODIM);
  combine_kernel<<<(NTOK * (ODIM / 4)) / 256, 256, 0, stream>>>(yb, tok_slot, tok_w, out);
}
